// TransformerBlock_13649406066944
// MI455X (gfx1250) — compile-verified, matrix operand order checked by reading, not run on hardware
//
#include <hip/hip_runtime.h>
#include <math.h>

#define E_DIM 1024
#define H_NUM 16
#define HD_DIM 64
#define WIN 256
#define B_NUM 8
#define S_LEN 1024

typedef __bf16 bf16_t;
typedef __attribute__((ext_vector_type(16))) __bf16 v16bf;
typedef __attribute__((ext_vector_type(8)))  float  v8f;

__device__ __forceinline__ v8f wmma_bf16(v16bf a, v16bf b, v8f c) {
  // D(16x16,f32) = A(16x32,bf16) * B(32x16,bf16) + C
  return __builtin_amdgcn_wmma_f32_16x16x32_bf16(false, a, false, b,
                                                 (short)0, c, false, false);
}

// Load a 16x32 bf16 WMMA A/B fragment from a row-major tile (16 rows x 32 K)
// with leading dimension ld, following the CDNA5 16-bit operand layout:
// lanes 0-15 hold rows 0-15 with K {0..7,16..23}; lanes 16-31 same rows with
// K {8..15,24..31}; two 16-bit K values per VGPR.
__device__ __forceinline__ v16bf load_frag(const bf16_t* __restrict__ p, int ld) {
  const int lane = threadIdx.x & 31;
  const bf16_t* row = p + (size_t)(lane & 15) * ld;
  const int kb = (lane >> 4) << 3;
  v16bf f;
#pragma unroll
  for (int j = 0; j < 8; ++j) {
    int k = (j < 4) ? (kb + 2 * j) : (16 + kb + 2 * (j - 4));
    f[2 * j]     = row[k];
    f[2 * j + 1] = row[k + 1];
  }
  return f;
}

// ---------------------------------------------------------------------------
// Tensor Data Mover tile staging (CDNA5). Guarded: falls back to batched
// register copies if the builtin is unavailable on this toolchain.
// ---------------------------------------------------------------------------
#if defined(__has_builtin)
#  if __has_builtin(__builtin_amdgcn_tensor_load_to_lds) && \
      __has_builtin(__builtin_amdgcn_s_wait_tensorcnt)
#    define USE_TDM 1
#  endif
#endif
#ifndef USE_TDM
#  define USE_TDM 0
#endif

#define LDA_S 72   // LDS row stride in bf16 elements (144B, 16B-aligned)

#if USE_TDM
typedef unsigned int v4u __attribute__((ext_vector_type(4)));
typedef int          v8i __attribute__((ext_vector_type(8)));
typedef int          v4i __attribute__((ext_vector_type(4)));

__device__ __forceinline__ unsigned lds_addr_of(const void* p) {
  return (unsigned)(size_t)(__attribute__((address_space(3))) const void*)p;
}

// DMA a [rows x 64] bf16 tile from a row-major [dim1 x dim0] bf16 tensor into
// LDS with an 8-element row pad (32 DWORDs data + 4 DWORDs pad = stride 72).
__device__ __forceinline__ void tdm_load_tile(const bf16_t* gsrc, unsigned lds_byte,
                                              int rows, int dim0, int dim1) {
  const unsigned long long ga = (unsigned long long)(size_t)gsrc;
  v4u g0;
  g0[0] = 1u;                                    // count=1, user descriptor
  g0[1] = lds_byte;                              // lds_addr (bytes)
  g0[2] = (unsigned)(ga & 0xffffffffu);          // global_addr[31:0]
  g0[3] = (unsigned)((ga >> 32) & 0x01ffffffu)   // global_addr[56:32]
          | (2u << 30);                          // type = 2 ("image")
  v8i g1;
  g1[0] = (int)((1u << 16)                       // data_size = 2 bytes
               | (1u << 20)                      // pad_enable
               | (4u << 22)                      // pad_interval: 32 DWORDs
               | (3u << 25));                    // pad_amount: 4 DWORDs (8 bf16)
  g1[1] = (int)(((unsigned)dim0 & 0xffffu) << 16);         // tensor_dim0[15:0]
  g1[2] = (int)((((unsigned)dim0 >> 16) & 0xffffu)         // tensor_dim0[31:16]
               | (((unsigned)dim1 & 0xffffu) << 16));      // tensor_dim1[15:0]
  g1[3] = (int)((((unsigned)dim1 >> 16) & 0xffffu)         // tensor_dim1[31:16]
               | (64u << 16));                             // tile_dim0 = 64
  g1[4] = rows;                                 // tile_dim1 (tile_dim2 = 0)
  g1[5] = dim0;                                 // tensor_dim0_stride[31:0]
  g1[6] = 0;                                    // stride[47:32]=0, dim1_stride lo
  g1[7] = 0;
  v4i z = {0, 0, 0, 0};
#if __clang_major__ >= 23
  v8i z8 = {0, 0, 0, 0, 0, 0, 0, 0};
  __builtin_amdgcn_tensor_load_to_lds(g0, g1, z, z, z8, 0);
#else
  __builtin_amdgcn_tensor_load_to_lds(g0, g1, z, z, 0);
#endif
}
#endif // USE_TDM

// ---------------------------------------------------------------------------
// Generic bf16 GEMM:  Out = epilogue(A[M,K] @ W[N,K]^T + bias[N])
// Block tile 128x64, 8 waves (4x2), each wave a 32x32 output (4 accumulators),
// K staged in LDS (TDM when available) in steps of 64 -> 8 WMMAs per stage.
// mode 0: f32 store    mode 1: bf16 store
// mode 2: bf16 store of exact GELU   mode 4: bf16 store transposed [B][H][HD][S]
// ---------------------------------------------------------------------------
#define GT_M 128
#define GT_N 64
#define GT_K 64

__global__ __launch_bounds__(256) void gemm_bf16_kernel(
    const bf16_t* __restrict__ A, const bf16_t* __restrict__ W,
    const float* __restrict__ bias, void* __restrict__ Out,
    int M, int N, int K, int mode)
{
  __shared__ bf16_t As[GT_M * LDA_S];
  __shared__ bf16_t Bs[GT_N * LDA_S];
  const int tid  = threadIdx.x;
  const int lane = tid & 31;
  const int wave = tid >> 5;
  const int wm = wave >> 1;            // 0..3 -> 32-row M sub-tile
  const int wn = wave & 1;             // 0..1 -> 32-col N sub-tile
  const int m0 = blockIdx.y * GT_M;
  const int n0 = blockIdx.x * GT_N;

  v8f acc00 = {}, acc01 = {}, acc10 = {}, acc11 = {};

#if USE_TDM
  const unsigned asb = lds_addr_of(As);
  const unsigned bsb = lds_addr_of(Bs);
#else
  const int arow = tid >> 1;           // 0..127
  const int acol = (tid & 1) * 32;     // 0,32
  const int brow = tid >> 2;           // 0..63
  const int bcol = (tid & 3) * 16;     // 0,16,32,48
#endif

  for (int k0 = 0; k0 < K; k0 += GT_K) {
    __syncthreads();                   // previous stage fully consumed
#if USE_TDM
    if (wave == 0) {
      tdm_load_tile(&A[(size_t)m0 * K + k0], asb, GT_M, K, M);
      tdm_load_tile(&W[(size_t)n0 * K + k0], bsb, GT_N, K, N);
      __builtin_amdgcn_s_wait_tensorcnt(0);
    }
#else
    const bf16_t* ag = &A[(size_t)(m0 + arow) * K + k0 + acol];
    const bf16_t* wg = &W[(size_t)(n0 + brow) * K + k0 + bcol];
    uint4 ta[4], tb[2];
#pragma unroll
    for (int i = 0; i < 4; ++i) ta[i] = *(const uint4*)&ag[i * 8];
#pragma unroll
    for (int i = 0; i < 2; ++i) tb[i] = *(const uint4*)&wg[i * 8];
#pragma unroll
    for (int i = 0; i < 4; ++i)
      *(uint4*)&As[arow * LDA_S + acol + i * 8] = ta[i];
#pragma unroll
    for (int i = 0; i < 2; ++i)
      *(uint4*)&Bs[brow * LDA_S + bcol + i * 8] = tb[i];
    if (k0 + GT_K < K) {
      __builtin_prefetch(ag + GT_K, 0, 1);
      __builtin_prefetch(wg + GT_K, 0, 1);
    }
#endif
    __syncthreads();                   // stage visible to all waves
#pragma unroll
    for (int ks = 0; ks < GT_K; ks += 32) {
      v16bf a0 = load_frag(&As[(wm * 32) * LDA_S + ks], LDA_S);
      v16bf a1 = load_frag(&As[(wm * 32 + 16) * LDA_S + ks], LDA_S);
      v16bf b0 = load_frag(&Bs[(wn * 32) * LDA_S + ks], LDA_S);
      v16bf b1 = load_frag(&Bs[(wn * 32 + 16) * LDA_S + ks], LDA_S);
      acc00 = wmma_bf16(a0, b0, acc00);
      acc01 = wmma_bf16(a0, b1, acc01);
      acc10 = wmma_bf16(a1, b0, acc10);
      acc11 = wmma_bf16(a1, b1, acc11);
    }
  }

  const int hi = lane >> 4, nn = lane & 15;
  v8f accs[2][2] = {{acc00, acc01}, {acc10, acc11}};
#pragma unroll
  for (int i = 0; i < 2; ++i) {
#pragma unroll
    for (int j = 0; j < 2; ++j) {
#pragma unroll
      for (int r = 0; r < 8; ++r) {
        const int m = m0 + wm * 32 + i * 16 + r + 8 * hi; // C layout: VGPR r -> row r+8*hi
        const int c = n0 + wn * 32 + j * 16 + nn;
        const float y = accs[i][j][r] + bias[c];
        if (mode == 0) {
          ((float*)Out)[(size_t)m * N + c] = y;
        } else if (mode == 1) {
          ((bf16_t*)Out)[(size_t)m * N + c] = (bf16_t)y;
        } else if (mode == 2) {
          ((bf16_t*)Out)[(size_t)m * N + c] =
              (bf16_t)(0.5f * y * (1.0f + erff(y * 0.70710678118f)));
        } else { // mode 4: V stored transposed per head: [B][H][HD][S]
          const int b = m / S_LEN, s = m % S_LEN;
          const int h = c / HD_DIM, d = c % HD_DIM;
          ((bf16_t*)Out)[(((size_t)b * H_NUM + h) * HD_DIM + d) * S_LEN + s] = (bf16_t)y;
        }
      }
    }
  }
}

// ---------------------------------------------------------------------------
// Sliding-window flash attention. Block = 128 queries (8 waves x 16 rows),
// one (b,h) per block. K tile 32x64, V tile (pre-transposed) 64x32 in LDS.
// ---------------------------------------------------------------------------
#define LDK 72
#define LDV 40
#define LDP 40

__global__ __launch_bounds__(256) void attn_kernel(
    const bf16_t* __restrict__ Q, const bf16_t* __restrict__ Kc,
    const bf16_t* __restrict__ Vt, bf16_t* __restrict__ Ctx)
{
  __shared__ bf16_t Ks[32 * LDK];         // [key][hd]
  __shared__ bf16_t Vs[64 * LDV];         // [hd][key]
  __shared__ bf16_t Ps[8 * 16 * LDP];     // per-wave P bounce buffer
  const int tid  = threadIdx.x;
  const int lane = tid & 31;
  const int wave = tid >> 5;
  const int hi = lane >> 4, nn = lane & 15;
  const int b = blockIdx.z, h = blockIdx.y;
  const int q0blk = blockIdx.x * 128;
  const int q0w   = q0blk + wave * 16;

  const bf16_t* qbase = Q + ((size_t)b * S_LEN + q0w) * E_DIM + h * HD_DIM;
  const v16bf qa0 = load_frag(qbase, E_DIM);       // K = hd 0..31
  const v16bf qa1 = load_frag(qbase + 32, E_DIM);  // K = hd 32..63

  v8f c0 = {}, c1 = {}, c2 = {}, c3 = {};
  float mrow[8], lsum[8];
#pragma unroll
  for (int r = 0; r < 8; ++r) { mrow[r] = -1e30f; lsum[r] = 0.0f; }

  const int klo = max(0, q0blk - WIN);
  const int khi = min(S_LEN, q0blk + 127 + WIN + 1);
  const float scale = 0.125f; // 1/sqrt(64)

  const int kr = tid >> 3, kc = (tid & 7) * 8;   // K tile coop load (32x64)
  const int vr = tid >> 2, vc = (tid & 3) * 8;   // V tile coop load (64x32)
  const bf16_t* kbase = Kc + (size_t)b * S_LEN * E_DIM + h * HD_DIM;
  const bf16_t* vbase = Vt + ((size_t)b * H_NUM + h) * HD_DIM * S_LEN;
  bf16_t* Pw = &Ps[wave * 16 * LDP];

  for (int kv0 = klo; kv0 < khi; kv0 += 32) {
    __syncthreads();
    const uint4 tk = *(const uint4*)&kbase[(size_t)(kv0 + kr) * E_DIM + kc];
    const uint4 tv = *(const uint4*)&vbase[(size_t)vr * S_LEN + kv0 + vc];
    *(uint4*)&Ks[kr * LDK + kc] = tk;
    *(uint4*)&Vs[vr * LDV + vc] = tv;
    __syncthreads();

    const bool active = (kv0 <= q0w + 15 + WIN) && (kv0 + 31 >= q0w - WIN);
    if (!active) continue;   // uniform per wave; barriers already passed

    v8f s0 = {}, s1 = {};
    {
      v16bf k00 = load_frag(&Ks[0], LDK);              // keys 0-15, hd 0..31
      v16bf k01 = load_frag(&Ks[32], LDK);             // keys 0-15, hd 32..63
      s0 = wmma_bf16(qa0, k00, s0);
      s0 = wmma_bf16(qa1, k01, s0);
      v16bf k10 = load_frag(&Ks[16 * LDK], LDK);       // keys 16-31
      v16bf k11 = load_frag(&Ks[16 * LDK + 32], LDK);
      s1 = wmma_bf16(qa0, k10, s1);
      s1 = wmma_bf16(qa1, k11, s1);
    }

#pragma unroll
    for (int r = 0; r < 8; ++r) {
      const int qi = q0w + r + 8 * hi;
      const int j0 = kv0 + nn, j1 = kv0 + 16 + nn;
      float a  = s0[r] * scale;
      float bb = s1[r] * scale;
      if (qi - j0 > WIN || j0 - qi > WIN) a  = -1e30f;
      if (qi - j1 > WIN || j1 - qi > WIN) bb = -1e30f;
      float t = fmaxf(a, bb);
#pragma unroll
      for (int off = 1; off < 16; off <<= 1) t = fmaxf(t, __shfl_xor(t, off, 32));
      const float mn = fmaxf(mrow[r], t);
      const float sc = __expf(mrow[r] - mn);
      const float p0 = __expf(a - mn);
      const float p1 = __expf(bb - mn);
      float rs = p0 + p1;
#pragma unroll
      for (int off = 1; off < 16; off <<= 1) rs += __shfl_xor(rs, off, 32);
      lsum[r] = lsum[r] * sc + rs;
      mrow[r] = mn;
      c0[r] *= sc; c1[r] *= sc; c2[r] *= sc; c3[r] *= sc;
      Pw[(r + 8 * hi) * LDP + nn]      = (bf16_t)p0;
      Pw[(r + 8 * hi) * LDP + 16 + nn] = (bf16_t)p1;
    }

    const v16bf pa = load_frag(Pw, LDP);               // 16 queries x 32 keys
    const v16bf v0 = load_frag(&Vs[0], LDV);           // hd 0-15 x 32 keys
    const v16bf v1 = load_frag(&Vs[16 * LDV], LDV);
    const v16bf v2 = load_frag(&Vs[32 * LDV], LDV);
    const v16bf v3 = load_frag(&Vs[48 * LDV], LDV);
    c0 = wmma_bf16(pa, v0, c0);
    c1 = wmma_bf16(pa, v1, c1);
    c2 = wmma_bf16(pa, v2, c2);
    c3 = wmma_bf16(pa, v3, c3);
  }

  bf16_t* obase = Ctx + ((size_t)b * S_LEN + q0w) * E_DIM + h * HD_DIM;
#pragma unroll
  for (int r = 0; r < 8; ++r) {
    const float inv = 1.0f / lsum[r];
    const size_t ro = (size_t)(r + 8 * hi) * E_DIM;
    obase[ro + nn]      = (bf16_t)(c0[r] * inv);
    obase[ro + 16 + nn] = (bf16_t)(c1[r] * inv);
    obase[ro + 32 + nn] = (bf16_t)(c2[r] * inv);
    obase[ro + 48 + nn] = (bf16_t)(c3[r] * inv);
  }
}

// ---------------------------------------------------------------------------
// RoPE on raw x -> bf16 qk_in, plus bf16 cast of x for the V projection.
// ---------------------------------------------------------------------------
__global__ __launch_bounds__(256) void rope_kernel(
    const float* __restrict__ x, bf16_t* __restrict__ qk, bf16_t* __restrict__ xv)
{
  const size_t idx = (size_t)blockIdx.x * 256 + threadIdx.x;  // over B*S*E
  const int e = (int)(idx & (E_DIM - 1));
  const int s = (int)((idx >> 10) & (S_LEN - 1));
  const int d = e & 63;
  const int dm = d & 31;
  const float freq = powf(10000.0f, -(float)dm / 32.0f);
  const float th = (float)s * freq;
  const float cs = cosf(th), sn = sinf(th);
  const float xi = x[idx];
  const float rot = (d < 32) ? -x[idx + 32] : x[idx - 32];
  qk[idx] = (bf16_t)(xi * cs + rot * sn);
  xv[idx] = (bf16_t)xi;
}

__global__ __launch_bounds__(256) void f2bf_kernel(
    const float* __restrict__ src, bf16_t* __restrict__ dst, size_t n)
{
  const size_t i = (size_t)blockIdx.x * 256 + threadIdx.x;
  if (i < n) dst[i] = (bf16_t)src[i];
}

// ---------------------------------------------------------------------------
// y = LayerNorm(X + R) * g + b ; writes f32 and optional bf16 copy.
// One block per row of E=1024.
// ---------------------------------------------------------------------------
__global__ __launch_bounds__(256) void add_ln_kernel(
    const float* __restrict__ X, const float* __restrict__ R,
    const float* __restrict__ g, const float* __restrict__ bta,
    float* __restrict__ Yf, bf16_t* __restrict__ Yb)
{
  __shared__ float rs[8], rq[8];
  const int tid = threadIdx.x, lane = tid & 31, wave = tid >> 5;
  const size_t base = (size_t)blockIdx.x * E_DIM;
  float v[4], s = 0.f, q = 0.f;
#pragma unroll
  for (int i = 0; i < 4; ++i) {
    const int c = tid + i * 256;
    const float t = X[base + c] + R[base + c];
    v[i] = t; s += t; q += t * t;
  }
#pragma unroll
  for (int off = 1; off < 32; off <<= 1) {
    s += __shfl_xor(s, off, 32);
    q += __shfl_xor(q, off, 32);
  }
  if (lane == 0) { rs[wave] = s; rq[wave] = q; }
  __syncthreads();
  float S = 0.f, Qs = 0.f;
#pragma unroll
  for (int w = 0; w < 8; ++w) { S += rs[w]; Qs += rq[w]; }
  const float mu = S * (1.0f / E_DIM);
  const float var = Qs * (1.0f / E_DIM) - mu * mu;
  const float rstd = rsqrtf(var + 1e-5f);
#pragma unroll
  for (int i = 0; i < 4; ++i) {
    const int c = tid + i * 256;
    const float y = (v[i] - mu) * rstd * g[c] + bta[c];
    Yf[base + c] = y;
    if (Yb) Yb[base + c] = (bf16_t)y;
  }
}

// ---------------------------------------------------------------------------
extern "C" void kernel_launch(void* const* d_in, const int* in_sizes, int n_in,
                              void* d_out, int out_size, void* d_ws, size_t ws_size,
                              hipStream_t stream)
{
  (void)in_sizes; (void)n_in; (void)out_size; (void)ws_size;
  const float* x     = (const float*)d_in[0];
  const float* in_w  = (const float*)d_in[1];
  const float* in_b  = (const float*)d_in[2];
  const float* out_w = (const float*)d_in[3];
  const float* out_b = (const float*)d_in[4];
  const float* ln1g  = (const float*)d_in[5];
  const float* ln1b  = (const float*)d_in[6];
  const float* w1    = (const float*)d_in[7];
  const float* b1    = (const float*)d_in[8];
  const float* w2    = (const float*)d_in[9];
  const float* b2    = (const float*)d_in[10];
  const float* ln2g  = (const float*)d_in[11];
  const float* ln2b  = (const float*)d_in[12];

  const size_t NR = (size_t)B_NUM * S_LEN;   // 8192 rows
  const size_t EE = (size_t)E_DIM * E_DIM;   // 1M

  size_t off = 0;
  auto alloc = [&](size_t bytes) -> void* {
    void* p = (char*)d_ws + off;
    off += (bytes + 255) & ~(size_t)255;
    return p;
  };
  bf16_t* w_in_b = (bf16_t*)alloc(3 * EE * sizeof(bf16_t));
  bf16_t* w_o_b  = (bf16_t*)alloc(EE * sizeof(bf16_t));
  bf16_t* w1_b   = (bf16_t*)alloc(4 * EE * sizeof(bf16_t));
  bf16_t* w2_b   = (bf16_t*)alloc(4 * EE * sizeof(bf16_t));
  bf16_t* qk_in  = (bf16_t*)alloc(NR * E_DIM * sizeof(bf16_t));
  bf16_t* xv     = (bf16_t*)alloc(NR * E_DIM * sizeof(bf16_t));
  bf16_t* qb     = (bf16_t*)alloc(NR * E_DIM * sizeof(bf16_t));
  bf16_t* kb     = (bf16_t*)alloc(NR * E_DIM * sizeof(bf16_t));
  bf16_t* vtb    = (bf16_t*)alloc(NR * E_DIM * sizeof(bf16_t));
  bf16_t* ctxb   = (bf16_t*)alloc(NR * E_DIM * sizeof(bf16_t));
  float*  x1f    = (float*) alloc(NR * E_DIM * sizeof(float));
  bf16_t* x1b    = (bf16_t*)alloc(NR * E_DIM * sizeof(bf16_t));
  bf16_t* h1b    = (bf16_t*)alloc(NR * 4 * E_DIM * sizeof(bf16_t));
  float*  outf   = (float*)d_out;   // also used as f32 scratch (attn-out / ffn-out)

  // Weights -> bf16 once per launch (deterministic, replay-safe).
  f2bf_kernel<<<(int)((3 * EE + 255) / 256), 256, 0, stream>>>(in_w, w_in_b, 3 * EE);
  f2bf_kernel<<<(int)((EE + 255) / 256), 256, 0, stream>>>(out_w, w_o_b, EE);
  f2bf_kernel<<<(int)((4 * EE + 255) / 256), 256, 0, stream>>>(w1, w1_b, 4 * EE);
  f2bf_kernel<<<(int)((4 * EE + 255) / 256), 256, 0, stream>>>(w2, w2_b, 4 * EE);

  rope_kernel<<<(int)(NR * E_DIM / 256), 256, 0, stream>>>(x, qk_in, xv);

  const dim3 gP(E_DIM / GT_N, (int)(NR / GT_M));
  gemm_bf16_kernel<<<gP, 256, 0, stream>>>(qk_in, w_in_b,          in_b,             qb,  (int)NR, E_DIM, E_DIM, 1);
  gemm_bf16_kernel<<<gP, 256, 0, stream>>>(qk_in, w_in_b + EE,     in_b + E_DIM,     kb,  (int)NR, E_DIM, E_DIM, 1);
  gemm_bf16_kernel<<<gP, 256, 0, stream>>>(xv,    w_in_b + 2 * EE, in_b + 2 * E_DIM, vtb, (int)NR, E_DIM, E_DIM, 4);

  const dim3 gA(S_LEN / 128, H_NUM, B_NUM);
  attn_kernel<<<gA, 256, 0, stream>>>(qb, kb, vtb, ctxb);

  gemm_bf16_kernel<<<gP, 256, 0, stream>>>(ctxb, w_o_b, out_b, outf, (int)NR, E_DIM, E_DIM, 0);
  add_ln_kernel<<<(int)NR, 256, 0, stream>>>(x, outf, ln1g, ln1b, x1f, x1b);

  const dim3 gF1(4 * E_DIM / GT_N, (int)(NR / GT_M));
  gemm_bf16_kernel<<<gF1, 256, 0, stream>>>(x1b, w1_b, b1, h1b, (int)NR, 4 * E_DIM, E_DIM, 2);
  gemm_bf16_kernel<<<gP, 256, 0, stream>>>(h1b, w2_b, b2, outf, (int)NR, E_DIM, 4 * E_DIM, 0);
  add_ln_kernel<<<(int)NR, 256, 0, stream>>>(x1f, outf, ln2g, ln2b, outf, nullptr);
}
